// SMPActor_22806276342283
// MI455X (gfx1250) — matrix operand order repaired and checked by reading, not compiled, not run on hardware
//
#include <hip/hip_runtime.h>
#include <hip/hip_bf16.h>
#include <cstdint>
#include <cstddef>

// ---------------------------------------------------------------------------
// SMP actor megakernel for gfx1250 (MI455X).
//  - f16 WMMA (V_WMMA_F32_16X16X32_F16), f32 accumulate
//  - one workgroup = 64 rows, entire 10-module network fused through LDS
//  - weights pre-transposed/converted to f16 [N][KP] in d_ws by prep kernels
//  - LDS footprint 156,672 B  ->  2 workgroups per 320KB WGP
// ---------------------------------------------------------------------------

typedef __attribute__((ext_vector_type(16))) _Float16 v16h;
typedef __attribute__((ext_vector_type(8)))  _Float16 v8h;
typedef __attribute__((ext_vector_type(8)))  float    v8f;

#define B_ROWS   131072        // 2^17
#define M_TILE   64
#define NWAVES   8             // 256 threads
#define LDA      520           // f16 stride of big activation buffers (conflict-free)
#define XS       72            // f16 stride of the small x-input buffer
#define FS       36            // f32 stride of the final-layer scratch
#define ALPHA_C  0.3f

// state -> input-column map for the batch-scrambling pack() in the reference
__constant__ int c_statemap[5][5] = {
    { 6,  7,  8, -1, -1},   // knee_l
    {11, 12, 13, -1, -1},   // knee_r
    { 4,  5, -1, -1, -1},   // hip_l
    { 9, 10, -1, -1, -1},   // hip_r
    { 0,  1, -1,  2,  3},   // head
};

struct SmpArgs {
    const float* inputs;
    const _Float16 *upW1, *upW2, *upW3;
    const _Float16 *actW1, *actW2, *actW3;
    const _Float16 *msgW1, *msgW2, *msgW3;
    const float *upB1, *upB2, *upB3;
    const float *actB1, *actB2, *actB3;
    const float *msgB1, *msgB2, *msgB3;
    const float *upLNg, *upLNb, *dnLNg, *dnLNb;
    float* out;
};

// -------------------------- fragment loaders -------------------------------

__device__ __forceinline__ v16h make16(v8h lo, v8h hi) {
    union { v16h v; v8h h[2]; } u;
    u.h[0] = lo; u.h[1] = hi;
    return u.v;
}

// A (16x32 f16, row-major source with stride S halves):
//   lane<16 : elems 0..7 = A[lane][k0..k0+7],   8..15 = A[lane][k0+16..k0+23]
//   lane>=16: elems 0..7 = A[lane-16][k0+8..15],8..15 = A[lane-16][k0+24..31]
__device__ __forceinline__ v16h load_a_frag(const _Float16* base, int S, int lane, int k0) {
    const int hf  = lane >> 4;
    const int row = lane & 15;
    const _Float16* p = base + row * S + k0 + hf * 8;
    return make16(*(const v8h*)p, *(const v8h*)(p + 16));
}

// B (32x16 f16) from WT[N][KP] (K-contiguous):
//   lane<16 : elems 0..7 = B[k0..k0+7][lane],     8..15 = B[k0+16..k0+23][lane]
//   lane>=16: elems 0..7 = B[k0+8..15][lane-16],  8..15 = B[k0+24..31][lane-16]
__device__ __forceinline__ v16h load_b_frag(const _Float16* __restrict__ wt, int KP,
                                            int lane, int n0, int k0) {
    const int hf = lane >> 4;
    const int n  = n0 + (lane & 15);
    const _Float16* p = wt + (size_t)n * KP + k0 + hf * 8;
    return make16(*(const v8h*)p, *(const v8h*)(p + 16));
}

// ------------------------------- GEMM --------------------------------------
// 64 x NP = (64 x KP) @ (KP x NP), KP multiple of 32, NP multiple of 16.
// EPI==0: bias + leaky-relu -> f16 out16 (stride LDA)
// EPI==1: bias             -> f32 out32 (stride FS)
template<int EPI>
__device__ __forceinline__ void gemm(const _Float16* Ain, int AS,
                                     const _Float16* __restrict__ WT, int KP,
                                     const float* __restrict__ bias, int N, int NP,
                                     _Float16* out16, float* out32, int tid)
{
    const int wave = tid >> 5;
    const int lane = tid & 31;
    const int hf   = lane >> 4;
    const int ln15 = lane & 15;
    const int NT   = NP >> 4;

    for (int nt = wave; nt < NT; nt += NWAVES) {
        const int n0 = nt << 4;
        v8f acc[4] = {};
        for (int k0 = 0; k0 < KP; k0 += 32) {
            const v16h bm = load_b_frag(WT, KP, lane, n0, k0);
#pragma unroll
            for (int mi = 0; mi < 4; ++mi) {
                const v16h am = load_a_frag(Ain + mi * 16 * AS, AS, lane, k0);
                acc[mi] = __builtin_amdgcn_wmma_f32_16x16x32_f16(
                    false, am, false, bm, (short)0, acc[mi], false, false);
            }
        }
        const int n  = n0 + ln15;
        const float bn = (n < N) ? bias[n] : 0.f;
#pragma unroll
        for (int mi = 0; mi < 4; ++mi) {
#pragma unroll
            for (int v = 0; v < 8; ++v) {
                const int row = mi * 16 + v + 8 * hf;   // C/D layout: VGPR v -> M=v / v+8
                float val = acc[mi][v] + bn;
                if (EPI == 0) {
                    // leaky(x) = max(x, 0.3x)  (exact for 0 < alpha < 1)
                    val = fmaxf(val, ALPHA_C * val);
                    out16[row * LDA + n] = (_Float16)val;
                } else {
                    out32[row * FS + n] = val;
                }
            }
        }
    }
}

// LayerNorm over 14 columns of fscr (stride FS), scale/shift, write f16 slot[64][16]
__device__ __forceinline__ void ln14(const float* fscr, int off,
                                     const float* __restrict__ g, const float* __restrict__ bb,
                                     _Float16* dst, int tid)
{
    if (tid < M_TILE) {
        const float* row = fscr + tid * FS + off;
        float mu = 0.f;
#pragma unroll
        for (int j = 0; j < 14; ++j) mu += row[j];
        mu *= (1.f / 14.f);
        float var = 0.f;
#pragma unroll
        for (int j = 0; j < 14; ++j) { float d = row[j] - mu; var += d * d; }
        var *= (1.f / 14.f);
        const float rs = rsqrtf(var + 0.001f);
#pragma unroll
        for (int j = 0; j < 14; ++j)
            dst[tid * 16 + j] = (_Float16)((row[j] - mu) * rs * g[j] + bb[j]);
    }
}

// ----------------------------- modules -------------------------------------

__device__ void module_up(const SmpArgs& A, int sid,
                          const _Float16* m1, const _Float16* m2, _Float16* outslot,
                          _Float16* bufA, _Float16* bufB, _Float16* xbuf,
                          float* fscr, int r0, int tid)
{
    // build x = [state(5), m1(14), m2(14), pad..64)  (replicates the reference's
    // batch-interleaving pack(): element m of row r comes from column map[p] at
    // batch index (5r+m) mod B, with p = (5r+m) / B)
    for (int i = tid; i < M_TILE * 64; i += NWAVES * 32) {
        const int r = i >> 6, c = i & 63;
        float v = 0.f;
        if (c < 5) {
            const int t  = 5 * (r0 + r) + c;
            const int p  = t >> 17;              // / 131072
            const int ii = t & (B_ROWS - 1);     // % 131072
            const int col = c_statemap[sid][p];
            if (col >= 0) v = A.inputs[(size_t)ii * 24 + col];
        } else if (c < 19) {
            if (m1) v = (float)m1[r * 16 + (c - 5)];
        } else if (c < 33) {
            if (m2) v = (float)m2[r * 16 + (c - 19)];
        }
        xbuf[r * XS + c] = (_Float16)v;
    }
    __syncthreads();
    gemm<0>(xbuf, XS, A.upW1, 64, A.upB1, 512, 512, bufA, nullptr, tid);
    __syncthreads();
    gemm<0>(bufA, LDA, A.upW2, 512, A.upB2, 512, 512, bufB, nullptr, tid);
    __syncthreads();
    gemm<1>(bufB, LDA, A.upW3, 512, A.upB3, 14, 16, nullptr, fscr, tid);  // fscr aliases bufA (h1 dead)
    __syncthreads();
    ln14(fscr, 0, A.upLNg, A.upLNb, outslot, tid);
    __syncthreads();
}

__device__ void module_down(const SmpArgs& A, const _Float16* mup, const _Float16* mdn,
                            int mu_col, _Float16* m1out, _Float16* m2out,
                            _Float16* bufA, _Float16* bufB, _Float16* xbuf,
                            float* fscr, int r0, int tid)
{
    // x = [m_up(14), m_down(14), pad..32)
    for (int i = tid; i < M_TILE * 32; i += NWAVES * 32) {
        const int r = i >> 5, c = i & 31;
        float v = 0.f;
        if (c < 14)      v = (float)mup[r * 16 + c];
        else if (c < 28) { if (mdn) v = (float)mdn[r * 16 + (c - 14)]; }
        xbuf[r * XS + c] = (_Float16)v;
    }
    __syncthreads();

    if (mu_col >= 0) {   // act branch (only mu needed)
        gemm<0>(xbuf, XS, A.actW1, 32, A.actB1, 512, 512, bufA, nullptr, tid);
        __syncthreads();
        gemm<0>(bufA, LDA, A.actW2, 512, A.actB2, 512, 512, bufB, nullptr, tid);
        __syncthreads();
        gemm<1>(bufB, LDA, A.actW3, 512, A.actB3, 2, 16, nullptr, fscr, tid);
        __syncthreads();
        if (tid < M_TILE) A.out[(size_t)(r0 + tid) * 4 + mu_col] = fscr[tid * FS + 0];
        __syncthreads();
    }
    if (m1out || m2out) {   // msg branch
        gemm<0>(xbuf, XS, A.msgW1, 32, A.msgB1, 512, 512, bufA, nullptr, tid);
        __syncthreads();
        gemm<0>(bufA, LDA, A.msgW2, 512, A.msgB2, 512, 512, bufB, nullptr, tid);
        __syncthreads();
        gemm<1>(bufB, LDA, A.msgW3, 512, A.msgB3, 28, 32, nullptr, fscr, tid);
        __syncthreads();
        if (m1out) ln14(fscr, 0,  A.dnLNg, A.dnLNb, m1out, tid);
        if (m2out) ln14(fscr, 14, A.dnLNg, A.dnLNb, m2out, tid);
        __syncthreads();
    }
}

// ----------------------------- megakernel ----------------------------------

// dynamic LDS layout (halves):
//   bufA [64*520]  (f32 final-layer scratch 'fscr' aliases its first 9216 B)
//   bufB [64*520]
//   xbuf [64*72]
//   slots [7][64*16]   (dn_hip_l reuses up_head's slot; dn_hip_r reuses dn_head_l's)
constexpr size_t SMEM_HALVES =
    (size_t)M_TILE * LDA * 2 + (size_t)M_TILE * XS + 7ull * M_TILE * 16;
constexpr size_t SMEM_BYTES = SMEM_HALVES * 2;     // 156,672 B -> 2 WGs per WGP

__global__ __launch_bounds__(256) void smp_megakernel(SmpArgs A)
{
    extern __shared__ _Float16 smem[];
    _Float16* bufA  = smem;
    _Float16* bufB  = bufA + M_TILE * LDA;
    _Float16* xbuf  = bufB + M_TILE * LDA;
    _Float16* slots = xbuf + M_TILE * XS;
    float*    fscr  = (float*)bufA;                // alias: h1 is dead when fscr is live

    const int tid = threadIdx.x;
    const int r0  = blockIdx.x * M_TILE;

    _Float16* S_ukl   = slots + 0 * M_TILE * 16;
    _Float16* S_ukr   = slots + 1 * M_TILE * 16;
    _Float16* S_uhl   = slots + 2 * M_TILE * 16;
    _Float16* S_uhr   = slots + 3 * M_TILE * 16;
    _Float16* S_uhead = slots + 4 * M_TILE * 16;
    _Float16* S_dhl   = slots + 5 * M_TILE * 16;
    _Float16* S_dhr   = slots + 6 * M_TILE * 16;
    _Float16* S_dnl   = S_uhead;   // up_head dead after module 6's x-build
    _Float16* S_dnr   = S_dhl;     // dn_head_l dead after module 7's x-build

    // ---- upward pass ----
    module_up(A, 0, nullptr, nullptr, S_ukl,   bufA, bufB, xbuf, fscr, r0, tid);
    module_up(A, 1, nullptr, nullptr, S_ukr,   bufA, bufB, xbuf, fscr, r0, tid);
    module_up(A, 2, S_ukl,  nullptr, S_uhl,    bufA, bufB, xbuf, fscr, r0, tid);
    module_up(A, 3, S_ukr,  nullptr, S_uhr,    bufA, bufB, xbuf, fscr, r0, tid);
    module_up(A, 4, S_uhl,  S_uhr,   S_uhead,  bufA, bufB, xbuf, fscr, r0, tid);

    // ---- downward pass ----
    module_down(A, S_uhead, nullptr, -1, S_dhl, S_dhr,    bufA, bufB, xbuf, fscr, r0, tid);
    module_down(A, S_uhl,   S_dhl,    0, S_dnl, nullptr,  bufA, bufB, xbuf, fscr, r0, tid);
    module_down(A, S_uhr,   S_dhr,    2, nullptr, S_dnr,  bufA, bufB, xbuf, fscr, r0, tid);
    module_down(A, S_ukl,   S_dnl,    1, nullptr, nullptr, bufA, bufB, xbuf, fscr, r0, tid);
    module_down(A, S_ukr,   S_dnr,    3, nullptr, nullptr, bufA, bufB, xbuf, fscr, r0, tid);

    // sigma = tanh(1.0) constant
    {
        const size_t base = (size_t)B_ROWS * 4 + (size_t)r0 * 4;
        A.out[base + tid] = 0.761594155955764888f;   // 256 threads, 64 rows * 4 cols
    }
}

// --------------------- weight prep: f32 [K][N] -> f16 [NP][KP] -------------

__global__ void prep_wt(const float* __restrict__ W, _Float16* __restrict__ WT,
                        int K, int N, int KP, int NP)
{
    const int idx = blockIdx.x * blockDim.x + threadIdx.x;
    if (idx >= KP * NP) return;
    const int n = idx / KP;
    const int k = idx - n * KP;
    const float v = (k < K && n < N) ? W[(size_t)k * N + n] : 0.f;
    WT[(size_t)n * KP + k] = (_Float16)v;
}

// ------------------------------ launch -------------------------------------

enum : size_t {
    OFF_UPW1  = 0,                        // 512 x 64
    OFF_UPW2  = OFF_UPW1  + 512 * 64,     // 512 x 512
    OFF_UPW3  = OFF_UPW2  + 512 * 512,    //  16 x 512
    OFF_ACTW1 = OFF_UPW3  + 16 * 512,     // 512 x 32
    OFF_ACTW2 = OFF_ACTW1 + 512 * 32,     // 512 x 512
    OFF_ACTW3 = OFF_ACTW2 + 512 * 512,    //  16 x 512
    OFF_MSGW1 = OFF_ACTW3 + 16 * 512,     // 512 x 32
    OFF_MSGW2 = OFF_MSGW1 + 512 * 32,     // 512 x 512
    OFF_MSGW3 = OFF_MSGW2 + 512 * 512,    //  32 x 512
    WS_HALVES = OFF_MSGW3 + 32 * 512
};

extern "C" void kernel_launch(void* const* d_in, const int* in_sizes, int n_in,
                              void* d_out, int out_size, void* d_ws, size_t ws_size,
                              hipStream_t stream)
{
    (void)in_sizes; (void)n_in; (void)out_size; (void)ws_size;
    auto IN = [&](int i) { return (const float*)d_in[i]; };
    _Float16* ws = (_Float16*)d_ws;

    auto prep = [&](const float* W, size_t off, int K, int N, int KP, int NP) {
        const int total = KP * NP;
        prep_wt<<<(total + 255) / 256, 256, 0, stream>>>(W, ws + off, K, N, KP, NP);
    };
    // d_in order: inputs, up_d1{W,b}, up_d2{W,b}, up_d3{W,b}, up_ln{g,b},
    //             act_d1{W,b}, act_d2{W,b}, act_d3{W,b},
    //             msg_d1{W,b}, msg_d2{W,b}, msg_d3{W,b}, down_ln{g,b}
    prep(IN(1),  OFF_UPW1,  33,  512, 64,  512);
    prep(IN(3),  OFF_UPW2,  512, 512, 512, 512);
    prep(IN(5),  OFF_UPW3,  512, 14,  512, 16);
    prep(IN(9),  OFF_ACTW1, 28,  512, 32,  512);
    prep(IN(11), OFF_ACTW2, 512, 512, 512, 512);
    prep(IN(13), OFF_ACTW3, 512, 2,   512, 16);
    prep(IN(15), OFF_MSGW1, 28,  512, 32,  512);
    prep(IN(17), OFF_MSGW2, 512, 512, 512, 512);
    prep(IN(19), OFF_MSGW3, 512, 28,  512, 32);

    SmpArgs a;
    a.inputs = IN(0);
    a.upW1  = ws + OFF_UPW1;  a.upW2  = ws + OFF_UPW2;  a.upW3  = ws + OFF_UPW3;
    a.actW1 = ws + OFF_ACTW1; a.actW2 = ws + OFF_ACTW2; a.actW3 = ws + OFF_ACTW3;
    a.msgW1 = ws + OFF_MSGW1; a.msgW2 = ws + OFF_MSGW2; a.msgW3 = ws + OFF_MSGW3;
    a.upB1  = IN(2);  a.upB2  = IN(4);  a.upB3  = IN(6);
    a.upLNg = IN(7);  a.upLNb = IN(8);
    a.actB1 = IN(10); a.actB2 = IN(12); a.actB3 = IN(14);
    a.msgB1 = IN(16); a.msgB2 = IN(18); a.msgB3 = IN(20);
    a.dnLNg = IN(21); a.dnLNb = IN(22);
    a.out   = (float*)d_out;

    smp_megakernel<<<B_ROWS / M_TILE, NWAVES * 32, SMEM_BYTES, stream>>>(a);
}
